// QuantizedLinear_592705487116
// MI455X (gfx1250) — matrix operand order repaired and checked by reading, not compiled, not run on hardware
//
#include <hip/hip_runtime.h>
#include <hip/hip_bf16.h>

typedef __attribute__((ext_vector_type(16))) __bf16 v16bf;
typedef __attribute__((ext_vector_type(8)))  __bf16 v8bf;
typedef __attribute__((ext_vector_type(8)))  float  v8f;
typedef __attribute__((ext_vector_type(4)))  float  v4f;

#define BM 128
#define BN 128
#define BK 32
#define LDK 40                 // LDS row stride in bf16 elems: 80B, 16B-aligned, conflict-breaking
#define KDIM 4096
#define KTILES (KDIM / BK)

union V16U { v16bf v; v8bf h[2]; };

__device__ __forceinline__ void cvt_store16(__bf16* dst, v4f a, v4f b, v4f c, v4f d) {
    v8bf lo, hi;
    lo[0] = (__bf16)a[0]; lo[1] = (__bf16)a[1]; lo[2] = (__bf16)a[2]; lo[3] = (__bf16)a[3];
    lo[4] = (__bf16)b[0]; lo[5] = (__bf16)b[1]; lo[6] = (__bf16)b[2]; lo[7] = (__bf16)b[3];
    hi[0] = (__bf16)c[0]; hi[1] = (__bf16)c[1]; hi[2] = (__bf16)c[2]; hi[3] = (__bf16)c[3];
    hi[4] = (__bf16)d[0]; hi[5] = (__bf16)d[1]; hi[6] = (__bf16)d[2]; hi[7] = (__bf16)d[3];
    *(v8bf*)(dst)     = lo;
    *(v8bf*)(dst + 8) = hi;
}

__global__ __launch_bounds__(256)
void bitnet_wmma_bf16_gemm(const float* __restrict__ X,   // [M, K] f32
                           const float* __restrict__ W,   // [N, K] f32 (ternary values)
                           const float* __restrict__ Sc,  // scalar weight_scale
                           float* __restrict__ Out,       // [M, N] f32
                           int M, int N)
{
    __shared__ __attribute__((aligned(16))) __bf16 As[2][BM * LDK];
    __shared__ __attribute__((aligned(16))) __bf16 Bs[2][BN * LDK];
    const int bufElems = BM * LDK;

    const int tid  = threadIdx.x;
    const int lane = tid & 31;
    const int wid  = tid >> 5;
    const int wm   = wid >> 2;   // 0..1 : wave's 64-row slab
    const int wn   = wid & 3;    // 0..3 : wave's 32-col slab
    const int lrow = lane & 15;
    const int half = lane >> 4;

    const int mtile = blockIdx.y;
    const int ntile = blockIdx.x;

    // Staging assignment: each thread moves 16 f32 of A and 16 f32 of B per K-step.
    const int grow = tid >> 1;        // 0..127
    const int gcol = (tid & 1) << 4;  // 0 or 16
    const float* xg = X + (size_t)(mtile * BM + grow) * KDIM + gcol;
    const float* wg = W + (size_t)(ntile * BN + grow) * KDIM + gcol;
    __bf16* as_st = &As[0][0] + grow * LDK + gcol;
    __bf16* bs_st = &Bs[0][0] + grow * LDK + gcol;

    v8f acc[4][2];
    #pragma unroll
    for (int i = 0; i < 4; ++i)
        #pragma unroll
        for (int j = 0; j < 2; ++j)
            acc[i][j] = (v8f){0.f, 0.f, 0.f, 0.f, 0.f, 0.f, 0.f, 0.f};

    // ---- prologue: stage K-tile 0 into LDS buffer 0 ----
    {
        const v4f* xp = (const v4f*)xg;
        const v4f* wp = (const v4f*)wg;
        v4f x0 = xp[0], x1 = xp[1], x2 = xp[2], x3 = xp[3];
        v4f w0 = wp[0], w1 = wp[1], w2 = wp[2], w3 = wp[3];
        cvt_store16(as_st, x0, x1, x2, x3);
        cvt_store16(bs_st, w0, w1, w2, w3);
    }
    __syncthreads();

    #pragma unroll 1
    for (int kt = 0; kt < KTILES; ++kt) {
        const int cur = kt & 1;
        const bool haveNext = (kt + 1 < KTILES);

        // Issue next tile's global loads first so they overlap the WMMAs.
        v4f nx0, nx1, nx2, nx3, nw0, nw1, nw2, nw3;
        if (haveNext) {
            const v4f* xp = (const v4f*)(xg + (size_t)(kt + 1) * BK);
            const v4f* wp = (const v4f*)(wg + (size_t)(kt + 1) * BK);
            nx0 = xp[0]; nx1 = xp[1]; nx2 = xp[2]; nx3 = xp[3];
            nw0 = wp[0]; nw1 = wp[1]; nw2 = wp[2]; nw3 = wp[3];
        }

        // ---- load ALL fragments into distinct registers (lets DS loads pipeline,
        //      then 8 WMMAs issue back-to-back with a single dscnt wait) ----
        const __bf16* Abase = &As[0][0] + cur * bufElems;
        const __bf16* Bbase = &Bs[0][0] + cur * bufElems;

        V16U bfrag[2];
        #pragma unroll
        for (int ns = 0; ns < 2; ++ns) {
            // B: 32x16, lane%16 = N col, lane/16 selects K half; K contiguous.
            const __bf16* bp = Bbase + (wn * 32 + ns * 16 + lrow) * LDK + half * 16;
            bfrag[ns].h[0] = *(const v8bf*)(bp);
            bfrag[ns].h[1] = *(const v8bf*)(bp + 8);
        }
        V16U afrag[4];
        #pragma unroll
        for (int ms = 0; ms < 4; ++ms) {
            // A: 16x32; lane<16 -> K {0..15}; lane>=16 -> K {16..31}
            const __bf16* ap = Abase + (wm * 64 + ms * 16 + lrow) * LDK;
            afrag[ms].h[0] = *(const v8bf*)(ap + half * 16);
            afrag[ms].h[1] = *(const v8bf*)(ap + half * 16 + 8);
        }

        #pragma unroll
        for (int ms = 0; ms < 4; ++ms) {
            acc[ms][0] = __builtin_amdgcn_wmma_f32_16x16x32_bf16(
                false, afrag[ms].v, false, bfrag[0].v, (short)0, acc[ms][0], false, false);
            acc[ms][1] = __builtin_amdgcn_wmma_f32_16x16x32_bf16(
                false, afrag[ms].v, false, bfrag[1].v, (short)0, acc[ms][1], false, false);
        }

        // ---- stage next tile into LDS[cur^1] ----
        if (haveNext) {
            const int nxt = (cur ^ 1) * bufElems;
            cvt_store16(as_st + nxt, nx0, nx1, nx2, nx3);
            cvt_store16(bs_st + nxt, nw0, nw1, nw2, nw3);
        }
        __syncthreads();
    }

    // ---- epilogue: scale and store f32 ----
    const float scale = *Sc;
    #pragma unroll
    for (int ms = 0; ms < 4; ++ms) {
        const size_t rowbase = (size_t)(mtile * BM + wm * 64 + ms * 16 + half * 8);
        #pragma unroll
        for (int ns = 0; ns < 2; ++ns) {
            const int col = ntile * BN + wn * 32 + ns * 16 + lrow;
            #pragma unroll
            for (int r = 0; r < 8; ++r) {
                Out[(rowbase + r) * (size_t)N + col] = acc[ms][ns][r] * scale;
            }
        }
    }
}

extern "C" void kernel_launch(void* const* d_in, const int* in_sizes, int n_in,
                              void* d_out, int out_size, void* d_ws, size_t ws_size,
                              hipStream_t stream) {
    const float* x     = (const float*)d_in[0];   // [8,2048,4096] f32
    const float* qw    = (const float*)d_in[1];   // [4096,4096] f32 ternary
    const float* scale = (const float*)d_in[2];   // scalar
    float* out = (float*)d_out;

    const int K = KDIM;
    const int M = in_sizes[0] / K;   // 16384
    const int N = in_sizes[1] / K;   // 4096

    dim3 grid(N / BN, M / BM);       // x = N tiles (fast) -> A panel + full W stay in L2
    bitnet_wmma_bf16_gemm<<<grid, 256, 0, stream>>>(x, qw, scale, out, M, N);
}
